// Decoder_11639361372658
// MI455X (gfx1250) — compile-verified, matrix operand order checked by reading, not run on hardware
//
#include <hip/hip_runtime.h>
#include <cmath>

// ---------------------------------------------------------------------------
// Types for CDNA5 WMMA (wave32)
// ---------------------------------------------------------------------------
typedef _Float16 half_t;
typedef __attribute__((ext_vector_type(16))) _Float16 v16h;
typedef __attribute__((ext_vector_type(8)))  _Float16 v8h;
typedef __attribute__((ext_vector_type(8)))  float    v8f;

// Load one 16x32 f16 tile (rows r0..r0+15, k0..k0+31) from a row-major matrix
// with leading dimension ld, into the CDNA5 WMMA A/B register layout:
//   lanes 0-15  : row r0+lane,    halves 0-7 -> K k0+0..7,   halves 8-15 -> K k0+16..23
//   lanes 16-31 : row r0+lane-16, halves 0-7 -> K k0+8..15,  halves 8-15 -> K k0+24..31
// ld must be a multiple of 8 halves (16B-aligned rows); k0 multiple of 32.
__device__ __forceinline__ v16h load_tile16x32(const half_t* __restrict__ base,
                                               int ld, int r0, int k0, int lane) {
  const int hi = (lane >> 4) & 1;
  const half_t* p = base + (size_t)(r0 + (lane & 15)) * (size_t)ld + (size_t)k0 + hi * 8;
  v8h c0 = *reinterpret_cast<const v8h*>(p);
  v8h c1 = *reinterpret_cast<const v8h*>(p + 16);
  v16h r;
#pragma unroll
  for (int i = 0; i < 8; ++i) { r[i] = c0[i]; r[i + 8] = c1[i]; }
  return r;
}

__device__ __forceinline__ float sigmoidf_dev(float x) {
  return 1.0f / (1.0f + __expf(-x));
}

// ---------------------------------------------------------------------------
// WMMA GEMM: out[M,Ncols] = act( BETA*out + A[M,K] @ W[Ncols,K]^T + bias )
// A, W f16 row-major; out f32 with row stride ldo. One wave per block, each
// wave owns a 16 x (NT*16) output patch. NO runtime guards: M % 16 == 0,
// Ncols % (NT*16) == 0, K % 32 == 0 -- all guaranteed by launch parameters,
// so the k-loop is pure loads + v_wmma with accumulators pinned in VGPRs.
// ACT: 0 = none, 1 = tanh, 2 = sigmoid.
// ---------------------------------------------------------------------------
template <int NT, int BETA, int ACT>
__global__ __launch_bounds__(32)
void gemm_wmma_kernel(float* __restrict__ out,
                      const half_t* __restrict__ A,
                      const half_t* __restrict__ W,
                      const float* __restrict__ bias,
                      int K, int lda, int ldw, long ldo) {
  const int lane  = threadIdx.x;
  const int m0    = blockIdx.y * 16;
  const int nBase = blockIdx.x * (NT * 16);

  v8f acc[NT];
#pragma unroll
  for (int t = 0; t < NT; ++t) acc[t] = (v8f){};

  for (int k0 = 0; k0 < K; k0 += 32) {
    const v16h a = load_tile16x32(A, lda, m0, k0, lane);
#pragma unroll
    for (int t = 0; t < NT; ++t) {
      const v16h b = load_tile16x32(W, ldw, nBase + t * 16, k0, lane);
      acc[t] = __builtin_amdgcn_wmma_f32_16x16x32_f16(
          false, a, false, b, (short)0, acc[t], false, false);
    }
  }

  const int hi = lane >> 4;
  const int nl = lane & 15;
#pragma unroll
  for (int t = 0; t < NT; ++t) {
    const int n = nBase + t * 16 + nl;
    const float bv = bias[n];
#pragma unroll
    for (int v = 0; v < 8; ++v) {
      const long idx = (long)(m0 + v + hi * 8) * ldo + n;
      float r = acc[t][v] + bv;
      if (BETA) r += out[idx];
      if (ACT == 1)      r = tanhf(r);
      else if (ACT == 2) r = sigmoidf_dev(r);
      out[idx] = r;
    }
  }
}

// ---------------------------------------------------------------------------
// Elementwise / setup kernels
// ---------------------------------------------------------------------------
__global__ void cvt_f32_f16_kernel(half_t* __restrict__ dst,
                                   const float* __restrict__ src, long n) {
  long i = (long)blockIdx.x * blockDim.x + threadIdx.x;
  if (i < n) dst[i] = (half_t)src[i];
}

// avg16[b,e] = mean over N of img[b,n,e]
__global__ void avg_kernel(half_t* __restrict__ avg16,
                           const float* __restrict__ img, int Nn, int E) {
  int idx = blockIdx.x * blockDim.x + threadIdx.x;
  if (idx >= 32 * E) return;
  int b = idx / E, e = idx - b * E;
  float s = 0.f;
  const float* p = img + (size_t)b * Nn * E + e;
  for (int n = 0; n < Nn; ++n) s += p[(size_t)n * E];
  avg16[idx] = (half_t)(s / (float)Nn);
}

// emb16[b,t,h] = emb[captions[b,t], h] * sqrt(H)   (t < T)
__global__ void emb_gather_kernel(half_t* __restrict__ emb16,
                                  const float* __restrict__ emb,
                                  const int* __restrict__ captions,
                                  int T, int Tcap, int H, float scale) {
  long idx = (long)blockIdx.x * blockDim.x + threadIdx.x;
  long total = (long)32 * T * H;
  if (idx >= total) return;
  int b = (int)(idx / ((long)T * H));
  int r = (int)(idx - (long)b * T * H);
  int t = r / H, j = r - t * H;
  int tok = captions[b * Tcap + t];
  emb16[idx] = (half_t)(emb[(size_t)tok * H + j] * scale);
}

// e[row] = W_v . tanh(Ws[row,:] + hU[row/98,:]) + b_v    (one wave per row)
__global__ void att_e_kernel(float* __restrict__ ebuf,
                             const float* __restrict__ Ws,
                             const float* __restrict__ hU,
                             const float* __restrict__ Wv,
                             const float* __restrict__ bv,
                             int rows, int Nn, int H) {
  int warp = threadIdx.x >> 5;
  int lane = threadIdx.x & 31;
  int row = blockIdx.x * (blockDim.x >> 5) + warp;
  if (row >= rows) return;
  int b = row / Nn;
  const float* wr = Ws + (size_t)row * H;
  const float* hr = hU + (size_t)b * H;
  float partial = 0.f;
  for (int j = lane; j < H; j += 32)
    partial += Wv[j] * tanhf(wr[j] + hr[j]);
#pragma unroll
  for (int off = 16; off > 0; off >>= 1)
    partial += __shfl_xor(partial, off, 32);
  if (lane == 0) ebuf[row] = partial + bv[0];
}

// softmax over N per batch, then ctx[b,e] = sum_n alpha[n]*img[b,n,e]
__global__ void softmax_ctx_kernel(float* __restrict__ ctx,
                                   const float* __restrict__ ebuf,
                                   const float* __restrict__ img,
                                   int Nn, int E) {
  __shared__ float sal[128];
  int b = blockIdx.x, tid = threadIdx.x;
  if (tid < Nn) sal[tid] = ebuf[b * Nn + tid];
  __syncthreads();
  if (tid == 0) {
    float mx = sal[0];
    for (int n = 1; n < Nn; ++n) mx = fmaxf(mx, sal[n]);
    float s = 0.f;
    for (int n = 0; n < Nn; ++n) { float v = __expf(sal[n] - mx); sal[n] = v; s += v; }
    float inv = 1.f / s;
    for (int n = 0; n < Nn; ++n) sal[n] *= inv;
  }
  __syncthreads();
  const float* ib = img + (size_t)b * Nn * E;
  for (int e = tid; e < E; e += blockDim.x) {
    float s = 0.f;
    for (int n = 0; n < Nn; ++n) s += sal[n] * ib[(size_t)n * E + e];
    ctx[(size_t)b * E + e] = s;
  }
}

// x16[b,:] = [ emb16[b,t,:], gate[b,:]*ctx[b,:] ]
__global__ void build_x_kernel(half_t* __restrict__ x16,
                               const half_t* __restrict__ emb16,
                               const float* __restrict__ gate,
                               const float* __restrict__ ctx,
                               int t, int T, int H, int E) {
  int idx = blockIdx.x * blockDim.x + threadIdx.x;
  int total = 32 * (H + E);
  if (idx >= total) return;
  int b = idx / (H + E);
  int j = idx - b * (H + E);
  half_t v;
  if (j < H) v = emb16[((size_t)b * T + t) * H + j];
  else {
    int e = j - H;
    v = (half_t)(gate[(size_t)b * E + e] * ctx[(size_t)b * E + e]);
  }
  x16[idx] = v;
}

// LSTM cell update: gates layout [i f g o] along 4H; updates c (f32), h16 (f16)
__global__ void lstm_kernel(float* __restrict__ c, half_t* __restrict__ h16,
                            const float* __restrict__ gates, int H) {
  int idx = blockIdx.x * blockDim.x + threadIdx.x;
  if (idx >= 32 * H) return;
  int b = idx / H, j = idx - b * H;
  const float* g = gates + (size_t)b * 4 * H;
  float si = sigmoidf_dev(g[j]);
  float sf = sigmoidf_dev(g[H + j]);
  float gg = tanhf(g[2 * H + j]);
  float so = sigmoidf_dev(g[3 * H + j]);
  float c2 = sf * c[idx] + si * gg;
  c[idx] = c2;
  h16[idx] = (half_t)(so * tanhf(c2));
}

// ---------------------------------------------------------------------------
// Host launcher
// ---------------------------------------------------------------------------
extern "C" void kernel_launch(void* const* d_in, const int* in_sizes, int n_in,
                              void* d_out, int out_size, void* d_ws, size_t ws_size,
                              hipStream_t stream) {
  (void)in_sizes; (void)n_in; (void)out_size; (void)ws_size;

  const int Bm = 32, Nn = 98, E = 2048, H = 512, V = 10000, Tcap = 61, T = 60;

  const float* img       = (const float*)d_in[0];
  const int*   captions  = (const int*)  d_in[1];
  const float* emb       = (const float*)d_in[2];
  const float* W_init_h  = (const float*)d_in[3];  const float* b_init_h = (const float*)d_in[4];
  const float* W_init_c  = (const float*)d_in[5];  const float* b_init_c = (const float*)d_in[6];
  const float* W_U       = (const float*)d_in[7];  const float* b_U      = (const float*)d_in[8];
  const float* W_att     = (const float*)d_in[9];  const float* b_att    = (const float*)d_in[10];
  const float* W_v       = (const float*)d_in[11]; const float* b_v      = (const float*)d_in[12];
  const float* W_fb      = (const float*)d_in[13]; const float* b_fb     = (const float*)d_in[14];
  const float* W_ih      = (const float*)d_in[15]; const float* b_ih     = (const float*)d_in[16];
  const float* W_hh      = (const float*)d_in[17]; const float* b_hh     = (const float*)d_in[18];
  const float* W_out     = (const float*)d_in[19]; const float* b_out    = (const float*)d_in[20];
  float* outp = (float*)d_out;

  // --- workspace carve-out (256B-aligned) ---
  uint8_t* wsb = (uint8_t*)d_ws;
  size_t off = 0;
  auto alloc = [&](size_t bytes) -> void* {
    void* p = (void*)(wsb + off);
    off += (bytes + 255) & ~(size_t)255;
    return p;
  };
  half_t* W_att16  = (half_t*)alloc((size_t)H * E * 2);
  half_t* W_U16    = (half_t*)alloc((size_t)H * H * 2);
  half_t* W_fb16   = (half_t*)alloc((size_t)E * H * 2);
  half_t* W_ih16   = (half_t*)alloc((size_t)4 * H * (H + E) * 2);
  half_t* W_hh16   = (half_t*)alloc((size_t)4 * H * H * 2);
  half_t* W_out16  = (half_t*)alloc((size_t)V * H * 2);
  half_t* W_inith16= (half_t*)alloc((size_t)H * E * 2);
  half_t* W_initc16= (half_t*)alloc((size_t)H * E * 2);
  half_t* img16    = (half_t*)alloc((size_t)Bm * Nn * E * 2);
  half_t* avg16    = (half_t*)alloc((size_t)Bm * E * 2);
  half_t* emb16    = (half_t*)alloc((size_t)Bm * T * H * 2);
  float*  Ws       = (float*) alloc((size_t)Bm * Nn * H * 4);
  float*  hU       = (float*) alloc((size_t)Bm * H * 4);
  float*  h_f32    = (float*) alloc((size_t)Bm * H * 4);
  float*  cbuf     = (float*) alloc((size_t)Bm * H * 4);
  half_t* h16      = (half_t*)alloc((size_t)Bm * H * 2);
  float*  ebuf     = (float*) alloc((size_t)Bm * Nn * 4);
  float*  ctx      = (float*) alloc((size_t)Bm * E * 4);
  float*  gate     = (float*) alloc((size_t)Bm * E * 4);
  half_t* x16      = (half_t*)alloc((size_t)Bm * (H + E) * 2);
  float*  gates    = (float*) alloc((size_t)Bm * 4 * H * 4);

  auto cvt = [&](half_t* dst, const float* src, size_t n) {
    int blocks = (int)((n + 255) / 256);
    cvt_f32_f16_kernel<<<dim3(blocks), dim3(256), 0, stream>>>(dst, src, (long)n);
  };

  // Templated GEMM launch helpers. Requirements (all satisfied by our shapes):
  //   M % 16 == 0, K % 32 == 0, Ncols % (NT*16) == 0.
#define LAUNCH_GEMM(NT, BETA, ACT, o, A, Wt, bias, M, Ncol, K, lda, ldw, ldo)   \
  gemm_wmma_kernel<NT, BETA, ACT>                                               \
      <<<dim3((Ncol) / ((NT) * 16), (M) / 16), dim3(32), 0, stream>>>(          \
          (o), (A), (Wt), (bias), (K), (lda), (ldw), (long)(ldo))

  // --- one-time setup ---
  cvt(W_att16,   W_att,    (size_t)H * E);
  cvt(W_U16,     W_U,      (size_t)H * H);
  cvt(W_fb16,    W_fb,     (size_t)E * H);
  cvt(W_ih16,    W_ih,     (size_t)4 * H * (H + E));
  cvt(W_hh16,    W_hh,     (size_t)4 * H * H);
  cvt(W_out16,   W_out,    (size_t)V * H);
  cvt(W_inith16, W_init_h, (size_t)H * E);
  cvt(W_initc16, W_init_c, (size_t)H * E);
  cvt(img16,     img,      (size_t)Bm * Nn * E);

  avg_kernel<<<dim3((Bm * E + 255) / 256), dim3(256), 0, stream>>>(avg16, img, Nn, E);
  {
    long total = (long)Bm * T * H;
    emb_gather_kernel<<<dim3((int)((total + 255) / 256)), dim3(256), 0, stream>>>(
        emb16, emb, captions, T, Tcap, H, sqrtf((float)H));
  }

  // h0 = tanh(avg @ W_init_h^T + b), c0 = tanh(avg @ W_init_c^T + b)
  LAUNCH_GEMM(4, 0, 1, h_f32, avg16, W_inith16, b_init_h, Bm, H, E, E, E, H);
  LAUNCH_GEMM(4, 0, 1, cbuf,  avg16, W_initc16, b_init_c, Bm, H, E, E, E, H);
  cvt(h16, h_f32, (size_t)Bm * H);

  // Ws[b,n,:] = img @ W_att^T + b_att   (big one-time GEMM, M = 3136)
  LAUNCH_GEMM(4, 0, 0, Ws, img16, W_att16, b_att, Bm * Nn, H, E, E, E, H);

  // --- sequential decode: 60 timesteps ---
  const int rows = Bm * Nn;
  for (int t = 0; t < T; ++t) {
    // hU = h @ W_U^T + b_U
    LAUNCH_GEMM(4, 0, 0, hU, h16, W_U16, b_U, Bm, H, H, H, H, H);
    // attention energies
    att_e_kernel<<<dim3((rows + 7) / 8), dim3(256), 0, stream>>>(
        ebuf, Ws, hU, W_v, b_v, rows, Nn, H);
    // softmax + context
    softmax_ctx_kernel<<<dim3(Bm), dim3(256), 0, stream>>>(ctx, ebuf, img, Nn, E);
    // gate = sigmoid(h @ W_fb^T + b_fb)
    LAUNCH_GEMM(4, 0, 2, gate, h16, W_fb16, b_fb, Bm, E, H, H, H, E);
    // x = [emb_t, gate*context]
    build_x_kernel<<<dim3((Bm * (H + E) + 255) / 256), dim3(256), 0, stream>>>(
        x16, emb16, gate, ctx, t, T, H, E);
    // gates = x @ W_ih^T + b_ih + h @ W_hh^T + b_hh
    LAUNCH_GEMM(4, 0, 0, gates, x16, W_ih16, b_ih, Bm, 4 * H, H + E, H + E, H + E, 4 * H);
    LAUNCH_GEMM(4, 1, 0, gates, h16, W_hh16, b_hh, Bm, 4 * H, H,     H,     H,     4 * H);
    // LSTM cell
    lstm_kernel<<<dim3((Bm * H + 255) / 256), dim3(256), 0, stream>>>(cbuf, h16, gates, H);
    // pred[b,t,:] = h2 @ W_out^T + b_out  (direct into d_out, row stride T*V;
    // NT=5 -> 80-wide patches, 10000/80 = 125 exact)
    LAUNCH_GEMM(5, 0, 0, outp + (size_t)t * V, h16, W_out16, b_out, Bm, V, H, H, H, (long)T * V);
  }
#undef LAUNCH_GEMM
}
